// DropGIN_28123445854581
// MI455X (gfx1250) — matrix operand order repaired and checked by reading, not compiled, not run on hardware
//
#include <hip/hip_runtime.h>
#include <hip/hip_bf16.h>

#define N_NODES   20000
#define N_EDGES   320000
#define DIM       64
#define N_CLASSES 16
#define N_GRAPHS  128
#define N_RUNS    4
#define N_LAYERS  4
#define BN_EPS    1e-5f
#define RN        (N_RUNS * N_NODES)          /* 80000 rows for GEMMs   */
#define MTILES    (RN / 16)                   /* 5000 row tiles         */
#define HSZ       ((size_t)RN * DIM)          /* 5,120,000 floats       */
#define NP1       (N_NODES + 1)
#define NCH       ((NP1 + 255) / 256)         /* 79 scan chunks         */

typedef __attribute__((ext_vector_type(16))) __bf16 v16bf;
typedef __attribute__((ext_vector_type(8)))  float  v8f;

union Frag { v16bf v; unsigned u[8]; };

// round-to-nearest-even f32 -> bf16, packed pair into one dword
__device__ __forceinline__ unsigned bf16pair(float lo, float hi) {
    unsigned a = __float_as_uint(lo);
    unsigned b = __float_as_uint(hi);
    a = (a + 0x7FFFu + ((a >> 16) & 1u)) >> 16;
    b = (b + 0x7FFFu + ((b >> 16) & 1u)) >> 16;
    return (b << 16) | (a & 0xFFFFu);
}

// ---------------------------------------------------------------- utilities
__global__ void zero_f32(float* p, int n) {
    int t = blockIdx.x * 256 + threadIdx.x;
    if (t < n) p[t] = 0.f;
}
__global__ void zero_i32(int* p, int n) {
    int t = blockIdx.x * 256 + threadIdx.x;
    if (t < n) p[t] = 0;
}
__global__ void copy_i32(const int* __restrict__ a, int* __restrict__ b, int n) {
    int t = blockIdx.x * 256 + threadIdx.x;
    if (t < n) b[t] = a[t];
}

// ---------------------------------------------------------------- CSR build
__global__ void hist_kernel(const int* __restrict__ dst, int* __restrict__ deg, int e) {
    int t = blockIdx.x * 256 + threadIdx.x;
    if (t < e) atomicAdd(&deg[dst[t]], 1);
}

// in-place per-chunk exclusive scan; chunk sums to aux
__global__ __launch_bounds__(256) void scan1(int* data, int* aux, int n) {
    __shared__ int sm[256];
    int gid = blockIdx.x * 256 + threadIdx.x;
    int v = (gid < n) ? data[gid] : 0;
    sm[threadIdx.x] = v;
    __syncthreads();
    for (int off = 1; off < 256; off <<= 1) {
        int t = (threadIdx.x >= off) ? sm[threadIdx.x - off] : 0;
        __syncthreads();
        sm[threadIdx.x] += t;
        __syncthreads();
    }
    if (gid < n) data[gid] = sm[threadIdx.x] - v;   // exclusive
    if (threadIdx.x == 255) aux[blockIdx.x] = sm[255];
}

__global__ void scan2(int* aux, int n) {            // n <= 256
    __shared__ int sm[256];
    int t = threadIdx.x;
    sm[t] = (t < n) ? aux[t] : 0;
    __syncthreads();
    if (t == 0) {
        int run = 0;
        for (int i = 0; i < n; ++i) { int v = sm[i]; sm[i] = run; run += v; }
    }
    __syncthreads();
    if (t < n) aux[t] = sm[t];
}

__global__ void scan3(int* data, const int* __restrict__ aux, int n) {
    int gid = blockIdx.x * 256 + threadIdx.x;
    if (gid < n) data[gid] += aux[blockIdx.x];
}

__global__ void fill_kernel(const int* __restrict__ src, const int* __restrict__ dst,
                            int* __restrict__ cursor, int* __restrict__ srcs, int e) {
    int t = blockIdx.x * 256 + threadIdx.x;
    if (t < e) {
        int p = atomicAdd(&cursor[dst[t]], 1);
        srcs[p] = src[t];
    }
}

// ---------------------------------------------------------------- h0 init + pool layer 0 (float4)
__global__ __launch_bounds__(256)
void init_h(const float* __restrict__ x, const unsigned char* __restrict__ dmask,
            const int* __restrict__ batch, float* __restrict__ h,
            float* __restrict__ pooled0) {
    int t = blockIdx.x * 256 + threadIdx.x;         // N_NODES*16 items
    if (t >= N_NODES * 16) return;
    int n = t >> 4, d = (t & 15) << 2;
    const float4 xv = *(const float4*)(x + (size_t)n * DIM + d);
    float4 acc = make_float4(0.f, 0.f, 0.f, 0.f);
    #pragma unroll
    for (int r = 0; r < N_RUNS; ++r) {
        float m = dmask[r * N_NODES + n] ? 0.f : 1.f;
        float4 v = make_float4(xv.x * m, xv.y * m, xv.z * m, xv.w * m);
        *(float4*)(h + (size_t)r * N_NODES * DIM + (size_t)n * DIM + d) = v;
        acc.x += v.x; acc.y += v.y; acc.z += v.z; acc.w += v.w;
    }
    float* p = pooled0 + batch[n] * DIM + d;
    atomicAdd(p + 0, acc.x * (1.f / N_RUNS));
    atomicAdd(p + 1, acc.y * (1.f / N_RUNS));
    atomicAdd(p + 2, acc.z * (1.f / N_RUNS));
    atomicAdd(p + 3, acc.w * (1.f / N_RUNS));
}

// ---------------------------------------------------------------- z = h + sum_{neighbors} h  (gather, no atomics)
__global__ __launch_bounds__(256)
void agg_kernel(const float* __restrict__ h, const int* __restrict__ rowptr,
                const int* __restrict__ srcs, float* __restrict__ z) {
    int lane = threadIdx.x & 31;
    int gw = blockIdx.x * 8 + (threadIdx.x >> 5);
    if (gw >= RN) return;
    int n = gw >> 2, r = gw & 3;
    const float* hb = h + (size_t)r * N_NODES * DIM;
    const int d = lane << 1;                         // dims d, d+1 per lane
    float2 acc = *(const float2*)(hb + (size_t)n * DIM + d);
    int beg = rowptr[n], end = rowptr[n + 1];
    for (int j = beg; j < end; ++j) {
        int s = srcs[j];
        if (j + 1 < end) {                           // hide random-row latency
            int s2 = srcs[j + 1];
            __builtin_prefetch(hb + (size_t)s2 * DIM + d, 0, 2);
        }
        float2 nb = *(const float2*)(hb + (size_t)s * DIM + d);
        acc.x += nb.x;
        acc.y += nb.y;
    }
    *(float2*)(z + (size_t)r * N_NODES * DIM + (size_t)n * DIM + d) = acc;
}

// ---------------------------------------------------------------- WMMA GEMM: Y = act(X) @ W + bias, + BN stats
// One wave per 16-row tile; A fragments built once, reused across all 4 column tiles (8 WMMAs/wave).
// mode 0: act = identity ; mode 1: act = relu(x*snt[k] + snt[64+k]) (fused BN+ReLU of previous layer)
__global__ __launch_bounds__(256)
void gemm_bn(const float* __restrict__ X, const float* __restrict__ W,
             const float* __restrict__ bias, const float* __restrict__ snt,
             float* __restrict__ Y, float* __restrict__ stats, int mode) {
    const int lane = threadIdx.x & 31;
    const int mt   = blockIdx.x * 8 + (threadIdx.x >> 5);
    if (mt >= MTILES) return;
    const int half = lane >> 4;
    const int l15  = lane & 15;
    const int m0   = mt << 4;
    const float* xr = X + (size_t)(m0 + l15) * DIM;

    // ---- A fragments (16x32 bf16 each): lane needs 4 contiguous 8-float blocks
    Frag a0, a1;
    #pragma unroll
    for (int blk = 0; blk < 4; ++blk) {
        const int base = (blk << 4) + (half << 3);
        float4 q0 = *(const float4*)(xr + base);
        float4 q1 = *(const float4*)(xr + base + 4);
        if (mode) {
            q0.x = fmaxf(fmaf(q0.x, snt[base + 0], snt[64 + base + 0]), 0.f);
            q0.y = fmaxf(fmaf(q0.y, snt[base + 1], snt[64 + base + 1]), 0.f);
            q0.z = fmaxf(fmaf(q0.z, snt[base + 2], snt[64 + base + 2]), 0.f);
            q0.w = fmaxf(fmaf(q0.w, snt[base + 3], snt[64 + base + 3]), 0.f);
            q1.x = fmaxf(fmaf(q1.x, snt[base + 4], snt[64 + base + 4]), 0.f);
            q1.y = fmaxf(fmaf(q1.y, snt[base + 5], snt[64 + base + 5]), 0.f);
            q1.z = fmaxf(fmaf(q1.z, snt[base + 6], snt[64 + base + 6]), 0.f);
            q1.w = fmaxf(fmaf(q1.w, snt[base + 7], snt[64 + base + 7]), 0.f);
        }
        Frag& A = (blk & 2) ? a1 : a0;
        const int g = (blk & 1) << 2;
        A.u[g + 0] = bf16pair(q0.x, q0.y);
        A.u[g + 1] = bf16pair(q0.z, q0.w);
        A.u[g + 2] = bf16pair(q1.x, q1.y);
        A.u[g + 3] = bf16pair(q1.z, q1.w);
    }

    // ---- loop over the 4 output column tiles
    #pragma unroll
    for (int nt = 0; nt < 4; ++nt) {
        const int col = (nt << 4) + l15;
        Frag b0, b1;
        #pragma unroll
        for (int v = 0; v < 8; ++v) {
            const int kw = ((v >> 2) << 4) + (half << 3) + ((v & 3) << 1);   // B 32x16: lanes 0-15 K 0..7,16..23; 16-31 K 8..15,24..31
            b0.u[v] = bf16pair(W[kw * DIM + col],        W[(kw + 1) * DIM + col]);
            b1.u[v] = bf16pair(W[(32 + kw) * DIM + col], W[(33 + kw) * DIM + col]);
        }
        v8f acc = {};
        acc = __builtin_amdgcn_wmma_f32_16x16x32_bf16(false, a0.v, false, b0.v,
                                                      (short)0, acc, false, false);
        acc = __builtin_amdgcn_wmma_f32_16x16x32_bf16(false, a1.v, false, b1.v,
                                                      (short)0, acc, false, false);

        const float bv = bias[col];
        float s = 0.f, ss = 0.f;
        float* yr = Y + (size_t)(m0 + (half << 3)) * DIM + col;
        #pragma unroll
        for (int i = 0; i < 8; ++i) {
            float yv = acc[i] + bv;
            s  += yv;
            ss += yv * yv;
            yr[i * DIM] = yv;   // D layout: VGPR i -> row m0 + half*8 + i
        }
        s  += __shfl_xor(s, 16, 32);
        ss += __shfl_xor(ss, 16, 32);
        if (lane < 16) {
            atomicAdd(&stats[col], s);
            atomicAdd(&stats[DIM + col], ss);
        }
    }
}

// ---------------------------------------------------------------- BN stats -> per-channel scale/shift
__global__ void bn_reduce(const float* __restrict__ stats, const float* __restrict__ g,
                          const float* __restrict__ b, float* __restrict__ snt) {
    int c = threadIdx.x;
    if (c < DIM) {
        const float inv_cnt = 1.f / (float)RN;
        float m   = stats[c] * inv_cnt;
        float var = stats[DIM + c] * inv_cnt - m * m;
        float sc  = g[c] * rsqrtf(var + BN_EPS);
        snt[c]       = sc;
        snt[DIM + c] = b[c] - m * sc;
    }
}

// ---------------------------------------------------------------- h = relu(z*s+t), pool into layer (L+1) (float4)
__global__ __launch_bounds__(256)
void finalize(const float* __restrict__ zin, const float* __restrict__ snt,
              const int* __restrict__ batch, float* __restrict__ h,
              float* __restrict__ pooledL) {
    int t = blockIdx.x * 256 + threadIdx.x;         // N_NODES*16 items
    if (t >= N_NODES * 16) return;
    int n = t >> 4, d = (t & 15) << 2;
    const float4 sc = *(const float4*)(snt + d);
    const float4 sh = *(const float4*)(snt + DIM + d);
    float4 acc = make_float4(0.f, 0.f, 0.f, 0.f);
    #pragma unroll
    for (int r = 0; r < N_RUNS; ++r) {
        size_t idx = (size_t)r * N_NODES * DIM + (size_t)n * DIM + d;
        float4 zv = *(const float4*)(zin + idx);
        float4 v;
        v.x = fmaxf(fmaf(zv.x, sc.x, sh.x), 0.f);
        v.y = fmaxf(fmaf(zv.y, sc.y, sh.y), 0.f);
        v.z = fmaxf(fmaf(zv.z, sc.z, sh.z), 0.f);
        v.w = fmaxf(fmaf(zv.w, sc.w, sh.w), 0.f);
        *(float4*)(h + idx) = v;
        acc.x += v.x; acc.y += v.y; acc.z += v.z; acc.w += v.w;
    }
    float* p = pooledL + batch[n] * DIM + d;
    atomicAdd(p + 0, acc.x * (1.f / N_RUNS));
    atomicAdd(p + 1, acc.y * (1.f / N_RUNS));
    atomicAdd(p + 2, acc.z * (1.f / N_RUNS));
    atomicAdd(p + 3, acc.w * (1.f / N_RUNS));
}

// ---------------------------------------------------------------- readout: 5 tiny GEMVs + log_softmax
__global__ void readout(const float* __restrict__ pooled, const float* __restrict__ fcW,
                        const float* __restrict__ fcb, float* __restrict__ out) {
    int g = blockIdx.x;
    int lane = threadIdx.x;
    float logit = 0.f;
    if (lane < N_CLASSES) {
        for (int i = 0; i <= N_LAYERS; ++i) {
            const float* p = pooled + ((size_t)i * N_GRAPHS + g) * DIM;
            const float* w = fcW + (size_t)i * DIM * N_CLASSES + lane;
            float acc = fcb[i * N_CLASSES + lane];
            for (int d = 0; d < DIM; ++d) acc = fmaf(p[d], w[d * N_CLASSES], acc);
            logit += acc;
        }
    }
    float mx = (lane < N_CLASSES) ? logit : -3.4e38f;
    for (int off = 8; off >= 1; off >>= 1) mx = fmaxf(mx, __shfl_xor(mx, off, 32));
    float ex = (lane < N_CLASSES) ? expf(logit - mx) : 0.f;
    float sm = ex;
    for (int off = 8; off >= 1; off >>= 1) sm += __shfl_xor(sm, off, 32);
    if (lane < N_CLASSES) out[g * N_CLASSES + lane] = (logit - mx) - logf(sm);
}

// ================================================================ launch
extern "C" void kernel_launch(void* const* d_in, const int* in_sizes, int n_in,
                              void* d_out, int out_size, void* d_ws, size_t ws_size,
                              hipStream_t stream) {
    (void)in_sizes; (void)n_in; (void)out_size; (void)ws_size;

    const float*         x      = (const float*)d_in[0];
    const int*           ei     = (const int*)  d_in[1];
    const int*           batch  = (const int*)  d_in[2];
    const unsigned char* dmask  = (const unsigned char*)d_in[3];
    const float* convW1 = (const float*)d_in[4];
    const float* convb1 = (const float*)d_in[5];
    const float* mbn_g  = (const float*)d_in[6];
    const float* mbn_b  = (const float*)d_in[7];
    const float* convW2 = (const float*)d_in[8];
    const float* convb2 = (const float*)d_in[9];
    const float* bn_g   = (const float*)d_in[10];
    const float* bn_b   = (const float*)d_in[11];
    const float* fcW    = (const float*)d_in[12];
    const float* fcb    = (const float*)d_in[13];
    float* out = (float*)d_out;

    // workspace layout
    float* h      = (float*)d_ws;
    float* z      = h + HSZ;
    float* y      = z + HSZ;
    float* pooled = y + HSZ;                                   // (L+1)*128*64
    float* stats  = pooled + (size_t)(N_LAYERS + 1) * N_GRAPHS * DIM;
    float* snt    = stats + 2 * DIM;
    int*   rowptr = (int*)(snt + 2 * DIM);                     // NP1
    int*   cursor = rowptr + NP1;                              // N_NODES
    int*   aux    = cursor + N_NODES;                          // up to 128
    int*   srcs   = aux + 128;                                 // N_EDGES

    const int* esrc = ei;
    const int* edst = ei + N_EDGES;

    // ---- CSR build (dst-sorted adjacency), atomic-free aggregation thereafter
    zero_i32<<<(NP1 + 255) / 256, 256, 0, stream>>>(rowptr, NP1);
    hist_kernel<<<(N_EDGES + 255) / 256, 256, 0, stream>>>(edst, rowptr, N_EDGES);
    scan1<<<NCH, 256, 0, stream>>>(rowptr, aux, NP1);
    scan2<<<1, 256, 0, stream>>>(aux, NCH);
    scan3<<<NCH, 256, 0, stream>>>(rowptr, aux, NP1);
    copy_i32<<<(N_NODES + 255) / 256, 256, 0, stream>>>(rowptr, cursor, N_NODES);
    fill_kernel<<<(N_EDGES + 255) / 256, 256, 0, stream>>>(esrc, edst, cursor, srcs, N_EDGES);

    // ---- pooled accumulators + h0
    zero_f32<<<((N_LAYERS + 1) * N_GRAPHS * DIM + 255) / 256, 256, 0, stream>>>(
        pooled, (N_LAYERS + 1) * N_GRAPHS * DIM);
    init_h<<<(N_NODES * 16 + 255) / 256, 256, 0, stream>>>(x, dmask, batch, h, pooled);

    const int gemm_blocks = (MTILES + 7) / 8;   // 625 blocks of 8 waves
    for (int L = 0; L < N_LAYERS; ++L) {
        agg_kernel<<<RN / 8, 256, 0, stream>>>(h, rowptr, srcs, z);

        zero_f32<<<1, 128, 0, stream>>>(stats, 2 * DIM);
        gemm_bn<<<gemm_blocks, 256, 0, stream>>>(z, convW1 + (size_t)L * DIM * DIM,
                                                 convb1 + L * DIM, snt, y, stats, 0);
        bn_reduce<<<1, 64, 0, stream>>>(stats, mbn_g + L * DIM, mbn_b + L * DIM, snt);

        zero_f32<<<1, 128, 0, stream>>>(stats, 2 * DIM);
        gemm_bn<<<gemm_blocks, 256, 0, stream>>>(y, convW2 + (size_t)L * DIM * DIM,
                                                 convb2 + L * DIM, snt, z, stats, 1);
        bn_reduce<<<1, 64, 0, stream>>>(stats, bn_g + L * DIM, bn_b + L * DIM, snt);

        finalize<<<(N_NODES * 16 + 255) / 256, 256, 0, stream>>>(
            z, snt, batch, h, pooled + (size_t)(L + 1) * N_GRAPHS * DIM);
    }

    readout<<<N_GRAPHS, 32, 0, stream>>>(pooled, fcW, fcb, out);
}
